// MemoryEnhancedKMeans_49460843380951
// MI455X (gfx1250) — matrix operand order repaired and checked by reading, not compiled, not run on hardware
//
#include <hip/hip_runtime.h>
#include <hip/hip_bf16.h>

#define DD   256
#define KMEM 1024
#define SS   4096
#define NB   8
#define NC   10
#define NTOP 5

typedef __attribute__((ext_vector_type(16))) __bf16 v16bf;
typedef __attribute__((ext_vector_type(8)))  float  v8f;

union BF16x16 { v16bf v; unsigned int u[8]; };

__device__ __forceinline__ unsigned short f32_to_bf16(float f) {
    unsigned int u = __float_as_uint(f);
    u += 0x7FFFu + ((u >> 16) & 1u);
    return (unsigned short)(u >> 16);
}

// ---------------------------------------------------------------- convert memory to bf16
__global__ void k_cvt_mem(const float* __restrict__ mem, unsigned short* __restrict__ memb) {
    int i = blockIdx.x * 256 + threadIdx.x;   // grid covers KMEM*DD
    memb[i] = f32_to_bf16(mem[i]);
}

// ---------------------------------------------------------------- fused retrieval:
// sim = x . mem^T via bf16 WMMA (fp32 acc), running top-5 per row (packed keys),
// softmax over top-5, x_enh = x + sum attn*mem[idx]
__global__ __launch_bounds__(128) void k_retrieval(
        const float* __restrict__ x, const float* __restrict__ mem,
        const unsigned short* __restrict__ memb, float* __restrict__ xenh)
{
    __shared__ unsigned short lx[64 * 264];          // 64 rows bf16, padded stride
    __shared__ unsigned int   cand[4][16][80];       // per-wave top-5 candidates
    __shared__ float          sattn[4][16][NTOP];
    __shared__ int            sidx[4][16][NTOP];

    const int tid = threadIdx.x;
    const int wv  = tid >> 5;
    const int ln  = tid & 31;
    const int l15 = ln & 15;
    const int half = ln >> 4;
    const int rowBase = blockIdx.x * 64;

    // stage x tile (coalesced) -> bf16 LDS
    for (int e = tid; e < 64 * 256; e += 128) {
        int r = e >> 8, d = e & 255;
        lx[r * 264 + d] = f32_to_bf16(x[(size_t)(rowBase + r) * DD + d]);
    }
    __syncthreads();

    // A fragments for this wave's 16 rows, resident in registers (8 chunks of K=32)
    const int rloc = wv * 16 + l15;
    v16bf afrag[8];
#pragma unroll
    for (int c = 0; c < 8; ++c) {
        const unsigned int* p = (const unsigned int*)&lx[rloc * 264 + c * 32 + half * 16];
        const unsigned int* q = (const unsigned int*)&lx[rloc * 264 + c * 32 + half * 16 + 8];
        BF16x16 f;
        f.u[0] = p[0]; f.u[1] = p[1]; f.u[2] = p[2]; f.u[3] = p[3];
        f.u[4] = q[0]; f.u[5] = q[1]; f.u[6] = q[2]; f.u[7] = q[3];
        afrag[c] = f.v;
    }

    // per-lane running top-5 (sorted desc) for 8 rows, packed (value|col) keys
    unsigned int tk[8][NTOP];
#pragma unroll
    for (int r = 0; r < 8; ++r)
#pragma unroll
        for (int s = 0; s < NTOP; ++s) tk[r][s] = 0u;

    for (int t = 0; t < 64; ++t) {                    // 64 N-tiles of 16 columns
        const int ncol = t * 16 + l15;
        const unsigned short* bp = memb + (size_t)ncol * DD + half * 16;
        v8f acc = {0.f, 0.f, 0.f, 0.f, 0.f, 0.f, 0.f, 0.f};
#pragma unroll
        for (int c = 0; c < 8; ++c) {
            v16bf bf = *(const v16bf*)(bp + c * 32);
            acc = __builtin_amdgcn_wmma_f32_16x16x32_bf16(
                      false, afrag[c], false, bf, (short)0, acc, false, false);
        }
#pragma unroll
        for (int r = 0; r < 8; ++r) {
            unsigned int u = __float_as_uint(acc[r]);
            u ^= ((unsigned int)((int)u >> 31)) | 0x80000000u;   // order-preserving map
            unsigned int key = (u & 0xFFFFFC00u) | (unsigned int)ncol;
#pragma unroll
            for (int s = 0; s < NTOP; ++s) {                      // shift-insert (umax/umin)
                unsigned int mx = tk[r][s] > key ? tk[r][s] : key;
                unsigned int mn = tk[r][s] > key ? key : tk[r][s];
                tk[r][s] = mx; key = mn;
            }
        }
    }

    // dump candidates: row m = r + half*8 owned by the 16 lanes of that half
#pragma unroll
    for (int r = 0; r < 8; ++r)
#pragma unroll
        for (int s = 0; s < NTOP; ++s)
            cand[wv][r + half * 8][l15 * 5 + s] = tk[r][s];
    __syncthreads();

    if (ln < 16) {                                     // lane ln merges row ln of its wave
        unsigned int best[NTOP] = {0u, 0u, 0u, 0u, 0u};
        for (int j = 0; j < 80; ++j) {
            unsigned int key = cand[wv][ln][j];
#pragma unroll
            for (int s = 0; s < NTOP; ++s) {
                unsigned int mx = best[s] > key ? best[s] : key;
                unsigned int mn = best[s] > key ? key : best[s];
                best[s] = mx; key = mn;
            }
        }
        float vals[NTOP]; int idxs[NTOP];
#pragma unroll
        for (int s = 0; s < NTOP; ++s) {
            idxs[s] = (int)(best[s] & 1023u);
            unsigned int u = best[s] & 0xFFFFFC00u;
            u = (u & 0x80000000u) ? (u ^ 0x80000000u) : ~u;       // inverse order map
            vals[s] = __uint_as_float(u);
        }
        float m0 = vals[0], ssum = 0.f, ev[NTOP];
#pragma unroll
        for (int s = 0; s < NTOP; ++s) { ev[s] = __expf(vals[s] - m0); ssum += ev[s]; }
        float inv = 1.f / ssum;
#pragma unroll
        for (int s = 0; s < NTOP; ++s) {
            sattn[wv][ln][s] = ev[s] * inv;
            sidx [wv][ln][s] = idxs[s];
        }
    }
    __syncthreads();

    // x_enh = x + sum attn * mem[idx]  (fp32, coalesced)
    for (int e = ln; e < 16 * 256; e += 32) {
        int rr = e >> 8, d = e & 255;
        size_t g = (size_t)(rowBase + wv * 16 + rr) * DD + d;
        float a = x[g];
#pragma unroll
        for (int s = 0; s < NTOP; ++s)
            a = fmaf(sattn[wv][rr][s], mem[(size_t)sidx[wv][rr][s] * DD + d], a);
        xenh[g] = a;
    }
}

// ---------------------------------------------------------------- centroid init (gather)
__global__ void k_init(const float* __restrict__ xenh, const int* __restrict__ init_idx,
                       float* __restrict__ cent) {
    int bk = blockIdx.x;                  // 80 blocks = (b,k)
    int b = bk / NC;
    int s = init_idx[bk];
    cent[(size_t)bk * DD + threadIdx.x] = xenh[((size_t)b * SS + s) * DD + threadIdx.x];
}

// ---------------------------------------------------------------- assignment + partial sums
// block = 32 rows of one batch (4 waves x 8 rows); deterministic per-wave LDS accumulation
__global__ __launch_bounds__(128) void k_assign(
        const float* __restrict__ xenh, const float* __restrict__ cent,
        int* __restrict__ assign, float* __restrict__ sums_part,
        float* __restrict__ counts_part)
{
    __shared__ float cl[NC * DD];
    __shared__ float cn[NC];
    __shared__ float sw[4][NC * DD];
    __shared__ float cw[4][NC];

    const int tid = threadIdx.x, wv = tid >> 5, ln = tid & 31;
    const int b = blockIdx.x >> 7;       // 128 blocks per batch

    for (int i = tid; i < NC * DD; i += 128) cl[i] = cent[(size_t)b * NC * DD + i];
    for (int i = tid; i < 4 * NC * DD; i += 128) (&sw[0][0])[i] = 0.f;
    for (int i = tid; i < 4 * NC; i += 128) (&cw[0][0])[i] = 0.f;
    __syncthreads();
    if (tid < NC) {
        float a = 0.f;
        for (int d = 0; d < DD; ++d) a = fmaf(cl[tid * DD + d], cl[tid * DD + d], a);
        cn[tid] = a;
    }
    __syncthreads();

    for (int i = 0; i < 8; ++i) {
        size_t row = (size_t)blockIdx.x * 32 + wv * 8 + i;
        const float* xr = xenh + row * DD;
        float xv[8];
#pragma unroll
        for (int j = 0; j < 8; ++j) xv[j] = xr[ln + 32 * j];

        int best = 0; float bs = 3.402823e38f;
        for (int k = 0; k < NC; ++k) {
            float p = 0.f;
#pragma unroll
            for (int j = 0; j < 8; ++j) p = fmaf(xv[j], cl[k * DD + ln + 32 * j], p);
            for (int m = 1; m < 32; m <<= 1) p += __shfl_xor(p, m, 32);
            float sc = cn[k] - 2.f * p;
            if (sc < bs) { bs = sc; best = k; }
        }
        if (ln == 0) assign[row] = best;
#pragma unroll
        for (int j = 0; j < 8; ++j) sw[wv][best * DD + ln + 32 * j] += xv[j];
        if (ln == 0) cw[wv][best] += 1.f;
    }
    __syncthreads();

    for (int i = tid; i < NC * DD; i += 128)
        sums_part[(size_t)blockIdx.x * NC * DD + i] = sw[0][i] + sw[1][i] + sw[2][i] + sw[3][i];
    if (tid < NC)
        counts_part[(size_t)blockIdx.x * NC + tid] = cw[0][tid] + cw[1][tid] + cw[2][tid] + cw[3][tid];
}

// ---------------------------------------------------------------- centroid update (fixed-order reduce)
__global__ void k_update(const float* __restrict__ sums_part,
                         const float* __restrict__ counts_part, float* __restrict__ cent) {
    const int bk = blockIdx.x;            // 80 blocks = (b,k)
    const int b = bk / NC, k = bk - b * NC, d = threadIdx.x;
    __shared__ float cs;
    if (d == 0) {
        float c = 0.f;
        for (int j = 0; j < 128; ++j) c += counts_part[(size_t)(b * 128 + j) * NC + k];
        cs = c;
    }
    float a = 0.f;
    for (int j = 0; j < 128; ++j)
        a += sums_part[(size_t)(b * 128 + j) * NC * DD + k * DD + d];
    __syncthreads();
    cent[(size_t)bk * DD + d] = a / (cs + 1e-8f);
}

// ---------------------------------------------------------------- quantize output
__global__ void k_quant(const float* __restrict__ cent, const int* __restrict__ assign,
                        float* __restrict__ out) {
    size_t row = blockIdx.x;              // 32768 rows
    int b = (int)(row >> 12);
    int a = assign[row];
    out[row * DD + threadIdx.x] = cent[((size_t)b * NC + a) * DD + threadIdx.x];
}

// ----------------------------------------------------------------
extern "C" void kernel_launch(void* const* d_in, const int* in_sizes, int n_in,
                              void* d_out, int out_size, void* d_ws, size_t ws_size,
                              hipStream_t stream) {
    (void)in_sizes; (void)n_in; (void)out_size; (void)ws_size;
    const float* x    = (const float*)d_in[0];     // [8,4096,256]
    const float* mem  = (const float*)d_in[1];     // [1024,256]
    const int*   iidx = (const int*)d_in[2];       // [8,10]
    float* out = (float*)d_out;

    char* ws = (char*)d_ws;
    unsigned short* memb       = (unsigned short*)(ws);                 //   0.5 MB
    float*          xenh       = (float*)(ws + 524288);                 //  33.5 MB
    float*          cent       = (float*)(ws + 34078720);               //  80 KB
    float*          sums_part  = (float*)(ws + 34160640);               //  10.5 MB
    float*          counts_part= (float*)(ws + 44646400);               //  40 KB
    int*            assign     = (int*)(ws + 44687360);                 // 128 KB

    k_cvt_mem  <<<(KMEM * DD) / 256, 256, 0, stream>>>(mem, memb);
    k_retrieval<<<(NB * SS) / 64, 128, 0, stream>>>(x, mem, memb, xenh);
    k_init     <<<NB * NC, DD, 0, stream>>>(xenh, iidx, cent);

    for (int it = 0; it < 10; ++it) {
        k_assign <<<(NB * SS) / 32, 128, 0, stream>>>(xenh, cent, assign, sums_part, counts_part);
        k_update <<<NB * NC, DD, 0, stream>>>(sums_part, counts_part, cent);
    }
    k_quant<<<NB * SS, DD, 0, stream>>>(cent, assign, out);
}
